// Ponita_49100066128245
// MI455X (gfx1250) — matrix operand order repaired and checked by reading, not compiled, not run on hardware
//
#include <hip/hip_runtime.h>

// ---------------------------------------------------------------------------
// Ponita forward for MI455X (gfx1250, wave32, WMMA f16 16x16x32, f32 accum).
//
// Input order assumption (setup_inputs() insertion order, nested dicts in
// insertion order, layer list in order):
//  0 x[2048*16] 1 pos[2048*3] 2 edge_index[2*24576](i32) 3 batch[2048](i32)
//  4 basis.W1[128*14] 5 basis.b1 6 basis.W2[128*128] 7 basis.b2
//  8 fiber.W1[128*3]  9 fiber.b1 10 fiber.W2[128*128] 11 fiber.b2
// 12 embed_W[128*16]
// 13+15*l : Wk,Wf,Wkey,bkey,Wq,bq,conv_b,ln_g,ln_b,W1,b1,W2,b2,Wr,br
// ---------------------------------------------------------------------------

#define NN      2048
#define NE      24576
#define NO      20
#define HC      128
#define NB      128
#define NW      512
#define NLAY    2
#define NROW_E  (NE*NO)     // 491520
#define NROW_N  (NN*NO)     // 40960

typedef __attribute__((ext_vector_type(16))) _Float16 v16h;
typedef __attribute__((ext_vector_type(8)))  float    v8f;

// ---------------- WMMA helpers (CDNA5 16x16x32 f16, f32 accum) -------------

__device__ __forceinline__ v8f wmma16(v16h a, v16h b, v8f c) {
  return __builtin_amdgcn_wmma_f32_16x16x32_f16(false, a, false, b,
                                                (short)0, c, false, false);
}

// A fragment 16x32 from row-major f16 (global or LDS), row stride ld, K offset k0.
// Layout: lanes 0-15 row=lane, halves 0-15 -> K+0..15;
//         lanes 16-31 same rows, K base +16.
__device__ __forceinline__ v16h load_a_frag(const _Float16* A, int ld, int k0, int lane) {
  int r  = lane & 15;
  int ko = k0 + ((lane >> 4) << 4);
  const _Float16* p = A + (long)r * ld + ko;
  v16h a;
#pragma unroll
  for (int i = 0; i < 8; ++i) a[i] = p[i];
#pragma unroll
  for (int i = 0; i < 8; ++i) a[i + 8] = p[8 + i];
  return a;
}

// B fragment 32x16 where B[k][n] = W[n][k], W row-major [out][in] f16.
// Lanes 0-15: n=lane, K=k0..k0+15 ; lanes 16-31: same n, K=k0+16..k0+31.
__device__ __forceinline__ v16h load_b_frag(const _Float16* W, int n0, int ldw, int k0, int lane) {
  int n  = n0 + (lane & 15);
  int ko = k0 + ((lane >> 4) << 4);
  const _Float16* p = W + (long)n * ldw + ko;
  v16h b;
#pragma unroll
  for (int i = 0; i < 16; ++i) b[i] = p[i];
  return b;
}

// ---------------- misc device helpers --------------------------------------

__device__ __forceinline__ float gelu_f(float x) {
  return 0.5f * x * (1.0f + erff(x * 0.7071067811865476f));
}
__device__ __forceinline__ unsigned fenc(float f) {
  unsigned u = __float_as_uint(f);
  return (u >> 31) ? ~u : (u | 0x80000000u);
}
__device__ __forceinline__ float fdec(unsigned u) {
  return (u >> 31) ? __uint_as_float(u & 0x7fffffffu) : __uint_as_float(~u);
}
__device__ __forceinline__ void ori_vec(int o, float& ox, float& oy, float& oz) {
  const double PI = 3.14159265358979323846;
  double di = (double)o;
  double th = fmod(PI * di * (1.0 + sqrt(5.0)), 2.0 * PI);
  double ph = acos(1.0 - 2.0 * (di + 0.5) / (double)NO);
  ox = (float)(cos(th) * sin(ph));
  oy = (float)(sin(th) * sin(ph));
  oz = (float)cos(ph);
}

// ---------------- weight prep ----------------------------------------------

__global__ void cvt_f16_kernel(const float* __restrict__ s, _Float16* __restrict__ d, int n) {
  int i = blockIdx.x * blockDim.x + threadIdx.x;
  if (i < n) d[i] = (_Float16)s[i];
}
// basis W1 [128][14] -> zero-padded [128][32] f16
__global__ void pad_w1_kernel(const float* __restrict__ s, _Float16* __restrict__ d) {
  int i = blockIdx.x * blockDim.x + threadIdx.x;   // 4096
  int r = i >> 5, c = i & 31;
  d[i] = (c < 14) ? (_Float16)s[r * 14 + c] : (_Float16)0.0f;
}

// ---------------- fiber basis + per-layer fiber kernels (tiny) -------------

__global__ void fiber_kernel(const float* __restrict__ fW1, const float* __restrict__ fb1,
                             const float* __restrict__ fW2, const float* __restrict__ fb2,
                             const float* __restrict__ Wf0, const float* __restrict__ Wf1,
                             float* __restrict__ fk0, float* __restrict__ fk1) {
  __shared__ float ori[NO][3];
  __shared__ float h1[NB];
  __shared__ float fb[NB];
  int t = threadIdx.x;                       // 128
  if (t < NO) ori_vec(t, ori[t][0], ori[t][1], ori[t][2]);
  __syncthreads();
  int p = blockIdx.x / NO, o = blockIdx.x - p * NO;
  float d = ori[p][0]*ori[o][0] + ori[p][1]*ori[o][1] + ori[p][2]*ori[o][2];
  float a = fW1[t*3+0]*d + fW1[t*3+1]*(d*d) + fW1[t*3+2]*(d*d*d) + fb1[t];
  h1[t] = gelu_f(a);
  __syncthreads();
  float s = fb2[t];
  for (int k = 0; k < NB; ++k) s += fW2[t*NB+k] * h1[k];
  fb[t] = gelu_f(s);
  __syncthreads();
  float s0 = 0.f, s1 = 0.f;
  for (int b = 0; b < NB; ++b) { float v = fb[b]; s0 += Wf0[t*NB+b]*v; s1 += Wf1[t*NB+b]*v; }
  fk0[(p*NO+o)*HC + t] = s0;
  fk1[(p*NO+o)*HC + t] = s1;
}

// ---------------- node embedding -------------------------------------------

__global__ void embed_kernel(const float* __restrict__ x, const float* __restrict__ eW,
                             float* __restrict__ h) {
  int n = blockIdx.x, c = threadIdx.x;       // 128 threads
  float s = 0.f;
#pragma unroll
  for (int k = 0; k < 16; ++k) s += x[n*16+k] * eW[c*16+k];
  for (int o = 0; o < NO; ++o) h[((long)n*NO + o)*HC + c] = s;
}

// ---------------- kernel_basis MLP over E*O rows (WMMA) ---------------------

__global__ __launch_bounds__(256) void basis_kernel(
    const int* __restrict__ ei, const float* __restrict__ pos,
    const _Float16* __restrict__ W1p, const float* __restrict__ b1,
    const _Float16* __restrict__ W2h, const float* __restrict__ b2,
    _Float16* __restrict__ kb) {
  __shared__ _Float16 ldsA[8][16*32];
  __shared__ _Float16 ldsM[8][16*128];
  int wid = threadIdx.x >> 5, lane = threadIdx.x & 31;
  int base = (blockIdx.x * 8 + wid) * 16;    // row tile base in [0, E*O)
  if (lane < 16) {
    int row = base + lane;
    int e = row / NO, o = row - e * NO;
    int s = ei[e], r = ei[NE + e];
    float rx = pos[s*3+0]-pos[r*3+0], ry = pos[s*3+1]-pos[r*3+1], rz = pos[s*3+2]-pos[r*3+2];
    float ox, oy, oz; ori_vec(o, ox, oy, oz);
    float i1 = rx*ox + ry*oy + rz*oz;
    float px = rx - i1*ox, py = ry - i1*oy, pz = rz - i1*oz;
    float i2 = sqrtf(px*px + py*py + pz*pz);
    float f[14];
    f[0]=i1; f[1]=i2;
    f[2]=i1*i1; f[3]=i1*i2; f[4]=i2*i1; f[5]=i2*i2;
    f[6]=f[2]*i1; f[7]=f[2]*i2; f[8]=f[3]*i1; f[9]=f[3]*i2;
    f[10]=f[4]*i1; f[11]=f[4]*i2; f[12]=f[5]*i1; f[13]=f[5]*i2;
    _Float16* rl = &ldsA[wid][lane*32];
#pragma unroll
    for (int k = 0; k < 14; ++k) rl[k] = (_Float16)f[k];
#pragma unroll
    for (int k = 14; k < 32; ++k) rl[k] = (_Float16)0.0f;
  }
  __syncthreads();
  // GEMM1: [16,14(pad32)] x W1.T -> [16,128], GELU, into LDS (f16)
  v16h a0 = load_a_frag(&ldsA[wid][0], 32, 0, lane);
#pragma unroll
  for (int nt = 0; nt < 8; ++nt) {
    v16h b = load_b_frag(W1p, nt*16, 32, 0, lane);
    v8f acc = {};
    acc = wmma16(a0, b, acc);
    int n = nt*16 + (lane & 15);
    float bias = b1[n];
#pragma unroll
    for (int v = 0; v < 8; ++v) {
      int m = v + ((lane >> 4) << 3);
      ldsM[wid][m*128 + n] = (_Float16)gelu_f(acc[v] + bias);
    }
  }
  __syncthreads();
  // GEMM2: [16,128] x W2.T -> [16,128], GELU, store fp16 basis
  v16h af[4];
#pragma unroll
  for (int kc = 0; kc < 4; ++kc) af[kc] = load_a_frag(&ldsM[wid][0], 128, kc*32, lane);
#pragma unroll
  for (int nt = 0; nt < 8; ++nt) {
    v8f acc = {};
#pragma unroll
    for (int kc = 0; kc < 4; ++kc) {
      v16h b = load_b_frag(W2h, nt*16, 128, kc*32, lane);
      acc = wmma16(af[kc], b, acc);
    }
    int n = nt*16 + (lane & 15);
    float bias = b2[n];
#pragma unroll
    for (int v = 0; v < 8; ++v) {
      int m = v + ((lane >> 4) << 3);
      kb[(long)(base + m)*128 + n] = (_Float16)gelu_f(acc[v] + bias);
    }
  }
}

// ---------------- keys / queries GEMMs (per layer) --------------------------

__global__ __launch_bounds__(256) void kq_kernel(
    const float* __restrict__ h,
    const _Float16* __restrict__ Wkey, const float* __restrict__ bkey,
    const _Float16* __restrict__ Wq,   const float* __restrict__ bq,
    float* __restrict__ keys, float* __restrict__ qrs) {
  __shared__ _Float16 ldsA[8][16*128];
  int wid = threadIdx.x >> 5, lane = threadIdx.x & 31;
  int base = (blockIdx.x * 8 + wid) * 16;
  for (int idx = lane; idx < 16*128; idx += 32)
    ldsA[wid][idx] = (_Float16)h[(long)base*HC + idx];
  __syncthreads();
  v16h af[4];
#pragma unroll
  for (int kc = 0; kc < 4; ++kc) af[kc] = load_a_frag(&ldsA[wid][0], 128, kc*32, lane);
#pragma unroll
  for (int nt = 0; nt < 8; ++nt) {
    v8f ak = {}, aq = {};
#pragma unroll
    for (int kc = 0; kc < 4; ++kc) {
      v16h bk = load_b_frag(Wkey, nt*16, 128, kc*32, lane);
      ak = wmma16(af[kc], bk, ak);
      v16h bq_ = load_b_frag(Wq, nt*16, 128, kc*32, lane);
      aq = wmma16(af[kc], bq_, aq);
    }
    int n = nt*16 + (lane & 15);
    float bk = bkey[n], bqv = bq[n];
#pragma unroll
    for (int v = 0; v < 8; ++v) {
      int m = v + ((lane >> 4) << 3);
      keys[(long)(base + m)*HC + n] = ak[v] + bk;
      qrs [(long)(base + m)*HC + n] = aq[v] + bqv;
    }
  }
}

// ---------------- attention logits + global max ------------------------------

__global__ void logits_kernel(const int* __restrict__ ei,
                              const float* __restrict__ keys, const float* __restrict__ qrs,
                              float* __restrict__ logits, unsigned* __restrict__ gmax) {
  __shared__ unsigned red[256];
  int i = blockIdx.x * 256 + threadIdx.x;     // exactly E*O threads
  int e = i / NO, o = i - e * NO;
  int s = ei[e], r = ei[NE + e];
  const float* kp = &keys[(long)(s*NO + o)*HC];
  const float* qp = &qrs [(long)(r*NO + o)*HC];
  float acc = 0.f;
#pragma unroll 8
  for (int k = 0; k < HC; ++k) acc += kp[k] * qp[k];
  acc *= 0.08838834764831845f;                // 1/sqrt(128)
  logits[i] = acc;
  red[threadIdx.x] = fenc(acc);
  __syncthreads();
  for (int st = 128; st > 0; st >>= 1) {
    if (threadIdx.x < st) {
      unsigned a = red[threadIdx.x], b = red[threadIdx.x + st];
      red[threadIdx.x] = a > b ? a : b;
    }
    __syncthreads();
  }
  if (threadIdx.x == 0) atomicMax(gmax, red[0]);
}

__global__ void att_kernel(const int* __restrict__ ei, float* __restrict__ logits,
                           const unsigned* __restrict__ gmax, float* __restrict__ den) {
  int i = blockIdx.x * 256 + threadIdx.x;
  int e = i / NO, o = i - e * NO;
  float M = fdec(*gmax);
  float ev = expf(logits[i] - M);
  logits[i] = ev;                              // reuse buffer as e_att
  int r = ei[NE + e];
  atomicAdd(&den[r*NO + o], ev);
}

// ---------------- message: (basis @ Wk.T) * h[send] * att, scatter to x1 -----

__global__ __launch_bounds__(256) void msg_kernel(
    const int* __restrict__ ei, const _Float16* __restrict__ kb,
    const _Float16* __restrict__ Wk, const float* __restrict__ eatt,
    const float* __restrict__ den, const float* __restrict__ h,
    float* __restrict__ x1) {
  __shared__ int   shs[8][16];
  __shared__ int   shr[8][16];
  __shared__ float shw[8][16];
  int wid = threadIdx.x >> 5, lane = threadIdx.x & 31;
  int base = (blockIdx.x * 8 + wid) * 16;
  if (lane < 16) {
    int row = base + lane;
    int e = row / NO, o = row - e * NO;
    int s = ei[e], r = ei[NE + e];
    shs[wid][lane] = s*NO + o;
    shr[wid][lane] = r*NO + o;
    shw[wid][lane] = eatt[row] / (den[r*NO + o] + 1e-6f);
  }
  __syncthreads();
  v16h af[4];
#pragma unroll
  for (int kc = 0; kc < 4; ++kc) af[kc] = load_a_frag(&kb[(long)base*128], 128, kc*32, lane);
  int hrow[8], xrow[8]; float wv[8];
#pragma unroll
  for (int v = 0; v < 8; ++v) {
    int m = v + ((lane >> 4) << 3);
    hrow[v] = shs[wid][m]; xrow[v] = shr[wid][m]; wv[v] = shw[wid][m];
    __builtin_prefetch(&h[(long)hrow[v]*HC], 0, 0);   // global_prefetch
  }
#pragma unroll
  for (int nt = 0; nt < 8; ++nt) {
    v8f acc = {};
#pragma unroll
    for (int kc = 0; kc < 4; ++kc) {
      v16h b = load_b_frag(Wk, nt*16, 128, kc*32, lane);
      acc = wmma16(af[kc], b, acc);
    }
    int c = nt*16 + (lane & 15);
#pragma unroll
    for (int v = 0; v < 8; ++v) {
      float val = acc[v] * wv[v] * h[(long)hrow[v]*HC + c];
      atomicAdd(&x1[(long)xrow[v]*HC + c], val);
    }
  }
}

// ---------------- fiber mix (20x20) + LayerNorm -> xn (f16) -----------------

__global__ void mix_ln_kernel(const float* __restrict__ x1, const float* __restrict__ fk,
                              const float* __restrict__ conv_b,
                              const float* __restrict__ ln_g, const float* __restrict__ ln_b,
                              _Float16* __restrict__ xnh) {
  __shared__ float red[128];
  __shared__ float red2[128];
  int n = blockIdx.x / NO, p = blockIdx.x - n * NO, c = threadIdx.x;
  float acc = 0.f;
#pragma unroll 4
  for (int o = 0; o < NO; ++o)
    acc += x1[((long)n*NO + o)*HC + c] * fk[(p*NO + o)*HC + c];
  acc = acc * (1.0f / NO) + conv_b[c];
  red[c] = acc; red2[c] = acc * acc;
  __syncthreads();
  for (int st = 64; st > 0; st >>= 1) {
    if (c < st) { red[c] += red[c + st]; red2[c] += red2[c + st]; }
    __syncthreads();
  }
  float mu  = red[0] * (1.0f / HC);
  float var = red2[0] * (1.0f / HC) - mu * mu;         // biased, torch LN
  float xn  = (acc - mu) * rsqrtf(var + 1e-5f) * ln_g[c] + ln_b[c];
  xnh[((long)n*NO + p)*HC + c] = (_Float16)xn;
}

// ---------------- ConvNeXt FF (128->512->128) + residual --------------------

__global__ __launch_bounds__(128) void ff_kernel(
    const _Float16* __restrict__ xnh,
    const _Float16* __restrict__ W1h, const float* __restrict__ b1,
    const _Float16* __restrict__ W2h, const float* __restrict__ b2,
    float* __restrict__ h) {
  __shared__ _Float16 mid[4][16*512];        // 64 KB
  int wid = threadIdx.x >> 5, lane = threadIdx.x & 31;
  int base = (blockIdx.x * 4 + wid) * 16;
  v16h af[4];
#pragma unroll
  for (int kc = 0; kc < 4; ++kc) af[kc] = load_a_frag(&xnh[(long)base*HC], HC, kc*32, lane);
#pragma unroll
  for (int nt = 0; nt < 32; ++nt) {
    v8f acc = {};
#pragma unroll
    for (int kc = 0; kc < 4; ++kc) {
      v16h b = load_b_frag(W1h, nt*16, HC, kc*32, lane);
      acc = wmma16(af[kc], b, acc);
    }
    int n = nt*16 + (lane & 15);
    float bias = b1[n];
#pragma unroll
    for (int v = 0; v < 8; ++v) {
      int m = v + ((lane >> 4) << 3);
      mid[wid][m*512 + n] = (_Float16)gelu_f(acc[v] + bias);
    }
  }
  __syncthreads();
#pragma unroll
  for (int nt = 0; nt < 8; ++nt) {
    v8f acc = {};
#pragma unroll
    for (int kc = 0; kc < 16; ++kc) {
      v16h a = load_a_frag(&mid[wid][0], 512, kc*32, lane);
      v16h b = load_b_frag(W2h, nt*16, 512, kc*32, lane);
      acc = wmma16(a, b, acc);
    }
    int n = nt*16 + (lane & 15);
    float bias = b2[n];
#pragma unroll
    for (int v = 0; v < 8; ++v) {
      int m = v + ((lane >> 4) << 3);
      long idx = (long)(base + m)*HC + n;
      h[idx] = h[idx] + acc[v] + bias;       // residual h += ff
    }
  }
}

// ---------------- readout accumulation + graph pooling ----------------------

__global__ void readout_kernel(const float* __restrict__ h, const float* __restrict__ Wr,
                               const float* __restrict__ br, float* __restrict__ racc) {
  int wid = threadIdx.x >> 5, lane = threadIdx.x & 31;
  int row = blockIdx.x * 8 + wid;
  const float* hp = &h[(long)row*HC];
  float s = 0.f;
  for (int k = lane; k < HC; k += 32) s += hp[k] * Wr[k];
#pragma unroll
  for (int off = 16; off > 0; off >>= 1) s += __shfl_down(s, off, 32);
  if (lane == 0) racc[row] += s + br[0];
}

__global__ void pool_kernel(const float* __restrict__ racc, const int* __restrict__ batch,
                            float* __restrict__ out) {
  int n = blockIdx.x * 256 + threadIdx.x;
  if (n < NN) {
    float s = 0.f;
    for (int o = 0; o < NO; ++o) s += racc[n*NO + o];
    s *= 1.0f / (float)(NLAY * NO);          // avg over layers & orientations
    atomicAdd(&out[batch[n]], s);
  }
}

// ---------------- workspace layout ------------------------------------------

static constexpr size_t SZ_KB    = (size_t)NROW_E * HC * 2;    // 125,829,120 (f16, L2-resident)
static constexpr size_t SZ_NOC_F = (size_t)NROW_N * HC * 4;    //  20,971,520
static constexpr size_t OFF_KB   = 0;
static constexpr size_t OFF_H    = OFF_KB  + SZ_KB;
static constexpr size_t OFF_KEY  = OFF_H   + SZ_NOC_F;
static constexpr size_t OFF_Q    = OFF_KEY + SZ_NOC_F;
static constexpr size_t OFF_X1   = OFF_Q   + SZ_NOC_F;
static constexpr size_t OFF_XNH  = OFF_X1  + SZ_NOC_F;
static constexpr size_t OFF_LOG  = OFF_XNH + (size_t)NROW_N * HC * 2;
static constexpr size_t OFF_DEN  = OFF_LOG + (size_t)NROW_E * 4;
static constexpr size_t OFF_RACC = OFF_DEN + (size_t)NROW_N * 4;
static constexpr size_t OFF_FK0  = OFF_RACC + (size_t)NROW_N * 4;
static constexpr size_t OFF_FK1  = OFF_FK0 + (size_t)NO * NO * HC * 4;
static constexpr size_t OFF_GMAX = OFF_FK1 + (size_t)NO * NO * HC * 4;
static constexpr size_t OFF_W    = (OFF_GMAX + 4 + 255) & ~(size_t)255;
static constexpr size_t OFF_W1P  = OFF_W;                       // 128*32 f16
static constexpr size_t OFF_W2B  = OFF_W1P + 128*32*2;          // 128*128 f16
static constexpr size_t OFF_LW   = OFF_W2B + 128*128*2;
static constexpr size_t SZ_LW    = (3*128*128 + 2*512*128) * 2; // per-layer f16 pool

extern "C" void kernel_launch(void* const* d_in, const int* in_sizes, int n_in,
                              void* d_out, int out_size, void* d_ws, size_t ws_size,
                              hipStream_t stream) {
  (void)in_sizes; (void)n_in; (void)out_size; (void)ws_size;
  const float* X     = (const float*)d_in[0];
  const float* POS   = (const float*)d_in[1];
  const int*   EI    = (const int*)d_in[2];
  const int*   BATCH = (const int*)d_in[3];
  const float* bW1   = (const float*)d_in[4];
  const float* bb1   = (const float*)d_in[5];
  const float* bW2   = (const float*)d_in[6];
  const float* bb2   = (const float*)d_in[7];
  const float* fW1   = (const float*)d_in[8];
  const float* fb1   = (const float*)d_in[9];
  const float* fW2   = (const float*)d_in[10];
  const float* fb2   = (const float*)d_in[11];
  const float* embW  = (const float*)d_in[12];
  auto LP = [&](int l, int j) { return (const float*)d_in[13 + l*15 + j]; };
  // layer j: 0 Wk 1 Wf 2 Wkey 3 bkey 4 Wq 5 bq 6 conv_b 7 ln_g 8 ln_b
  //          9 W1 10 b1 11 W2 12 b2 13 Wr 14 br

  char* ws = (char*)d_ws;
  _Float16* kb   = (_Float16*)(ws + OFF_KB);
  float*    h    = (float*)   (ws + OFF_H);
  float*    keys = (float*)   (ws + OFF_KEY);
  float*    qrs  = (float*)   (ws + OFF_Q);
  float*    x1   = (float*)   (ws + OFF_X1);
  _Float16* xnh  = (_Float16*)(ws + OFF_XNH);
  float*    logi = (float*)   (ws + OFF_LOG);
  float*    den  = (float*)   (ws + OFF_DEN);
  float*    racc = (float*)   (ws + OFF_RACC);
  float*    fk0  = (float*)   (ws + OFF_FK0);
  float*    fk1  = (float*)   (ws + OFF_FK1);
  unsigned* gmax = (unsigned*)(ws + OFF_GMAX);
  _Float16* w1p  = (_Float16*)(ws + OFF_W1P);
  _Float16* w2b  = (_Float16*)(ws + OFF_W2B);
  _Float16* lw[NLAY][5];
  for (int l = 0; l < NLAY; ++l) {
    char* base = ws + OFF_LW + (size_t)l * SZ_LW;
    lw[l][0] = (_Float16*)(base);                     // Wk
    lw[l][1] = (_Float16*)(base + 32768);             // Wkey
    lw[l][2] = (_Float16*)(base + 65536);             // Wq
    lw[l][3] = (_Float16*)(base + 98304);             // W1 ff [512,128]
    lw[l][4] = (_Float16*)(base + 98304 + 131072);    // W2 ff [128,512]
  }

  // ---- weight prep (f16 copies) ----
  pad_w1_kernel<<<16, 256, 0, stream>>>(bW1, w1p);
  cvt_f16_kernel<<<64, 256, 0, stream>>>(bW2, w2b, 128*128);
  for (int l = 0; l < NLAY; ++l) {
    cvt_f16_kernel<<<64,  256, 0, stream>>>(LP(l,0),  lw[l][0], 128*128);
    cvt_f16_kernel<<<64,  256, 0, stream>>>(LP(l,2),  lw[l][1], 128*128);
    cvt_f16_kernel<<<64,  256, 0, stream>>>(LP(l,4),  lw[l][2], 128*128);
    cvt_f16_kernel<<<256, 256, 0, stream>>>(LP(l,9),  lw[l][3], 512*128);
    cvt_f16_kernel<<<256, 256, 0, stream>>>(LP(l,11), lw[l][4], 128*512);
  }

  // ---- orientation-space (fiber) basis and per-layer fiber kernels ----
  fiber_kernel<<<NO*NO, 128, 0, stream>>>(fW1, fb1, fW2, fb2, LP(0,1), LP(1,1), fk0, fk1);

  // ---- embed + lift ----
  embed_kernel<<<NN, 128, 0, stream>>>(X, embW, h);

  // ---- kernel_basis MLP over all (edge, orientation) rows ----
  basis_kernel<<<NROW_E/16/8, 256, 0, stream>>>(EI, POS, w1p, bb1, w2b, bb2, kb);

  hipMemsetAsync(racc, 0, (size_t)NROW_N * 4, stream);

  for (int l = 0; l < NLAY; ++l) {
    kq_kernel<<<NROW_N/16/8, 256, 0, stream>>>(h, lw[l][1], LP(l,3), lw[l][2], LP(l,5),
                                               keys, qrs);
    hipMemsetAsync(gmax, 0, 4, stream);                 // encoded -inf
    hipMemsetAsync(den, 0, (size_t)NROW_N * 4, stream);
    logits_kernel<<<NROW_E/256, 256, 0, stream>>>(EI, keys, qrs, logi, gmax);
    att_kernel<<<NROW_E/256, 256, 0, stream>>>(EI, logi, gmax, den);
    hipMemsetAsync(x1, 0, SZ_NOC_F, stream);
    msg_kernel<<<NROW_E/16/8, 256, 0, stream>>>(EI, kb, lw[l][0], logi, den, h, x1);
    mix_ln_kernel<<<NROW_N, 128, 0, stream>>>(x1, l ? fk1 : fk0, LP(l,6), LP(l,7), LP(l,8), xnh);
    ff_kernel<<<NROW_N/16/4, 128, 0, stream>>>(xnh, lw[l][3], LP(l,10), lw[l][4], LP(l,12), h);
    readout_kernel<<<NROW_N/8, 256, 0, stream>>>(h, LP(l,13), LP(l,14), racc);
  }

  hipMemsetAsync(d_out, 0, 16 * 4, stream);
  pool_kernel<<<NN/256, 256, 0, stream>>>(racc, BATCH, (float*)d_out);
}
